// MSA_67473936221011
// MI455X (gfx1250) — matrix operand order repaired and checked by reading, not compiled, not run on hardware
//
#include <hip/hip_runtime.h>

typedef __attribute__((ext_vector_type(16))) _Float16 v16h;
typedef __attribute__((ext_vector_type(8)))  _Float16 v8h;
typedef __attribute__((ext_vector_type(4)))  _Float16 v4h;
typedef __attribute__((ext_vector_type(8)))  float    v8f;

#define TT 16
#define PP 8
#define HH 8
#define SS 128            // sequence = T*P
#define DD 32             // head dim
#define CC 768            // 3*H*DD
#define REL_N 1984        // (2T-1)*P*P = 31*64

#define LOG2E 1.4426950408889634f

// A-fragment (16-bit 16x32): lane = row, contiguous K 0..15 / 16..31 per lane-half
__device__ __forceinline__ v16h load_a_frag(const _Float16* base) {
  return *(const v16h*)(base);
}

// B-fragment (16-bit 32x16): lane = column, contiguous K 0..15 / 16..31 per lane-half
__device__ __forceinline__ v16h load_b_frag(const _Float16* base) {
  return *(const v16h*)(base);
}

__global__ __launch_bounds__(256) void msa_fused_f16wmma(
    const float* __restrict__ x, const float* __restrict__ rel,
    float* __restrict__ out) {
  __shared__ alignas(16) _Float16 Qs[SS * DD];     // 8 KB, pre-scaled by H^-.5*log2e
  __shared__ alignas(16) _Float16 Ks[SS * DD];     // 8 KB
  __shared__ alignas(16) _Float16 VTs[DD * SS];    // 8 KB, transposed V
  __shared__ alignas(16) _Float16 Ps[SS * SS];     // 32 KB unnormalized exp2 probs
  __shared__ alignas(16) float rels[REL_N];        // 7.9 KB bias slice * log2e

  const int tid = threadIdx.x;
  const int b = blockIdx.x >> 3;
  const int h = blockIdx.x & 7;
  // H^-0.5 * log2(e): softmax computed in exp2 domain (invariant rescale)
  const float scale = 0.35355339059327373f * LOG2E;

  // ---- stage Q/K/V as f16 into LDS (float4 loads, packed f16 stores) ----
  const float* xb = x + (size_t)b * SS * CC + h * DD;
  for (int i = tid; i < SS * DD / 4; i += 256) {   // 1024 float4 groups
    int s = i >> 3;
    int d4 = (i & 7) * 4;
    const float* p = xb + (size_t)s * CC + d4;
    float4 q = *(const float4*)(p);
    float4 k = *(const float4*)(p + HH * DD);
    float4 v = *(const float4*)(p + 2 * HH * DD);
    v4h qh = {(_Float16)(q.x * scale), (_Float16)(q.y * scale),
              (_Float16)(q.z * scale), (_Float16)(q.w * scale)};
    v4h kh = {(_Float16)k.x, (_Float16)k.y, (_Float16)k.z, (_Float16)k.w};
    *(v4h*)&Qs[s * DD + d4] = qh;
    *(v4h*)&Ks[s * DD + d4] = kh;
    VTs[(d4 + 0) * SS + s] = (_Float16)v.x;
    VTs[(d4 + 1) * SS + s] = (_Float16)v.y;
    VTs[(d4 + 2) * SS + s] = (_Float16)v.z;
    VTs[(d4 + 3) * SS + s] = (_Float16)v.w;
  }
  for (int i = tid; i < REL_N; i += 256) rels[i] = rel[i * HH + h] * LOG2E;
  __syncthreads();

  const int w = tid >> 5;      // wave id: owns s-columns [16w, 16w+16)
  const int lane = tid & 31;
  const int ln = lane & 15;
  const int hb = lane >> 4;
  const int kb = hb * 8;

  const int s_me = 16 * w + ln;        // this lane's s (N dim of S^T tiles)
  const int ti_s = s_me >> 3;          // temporal index of s
  const int pi = s_me & 7;             // spatial index of s

  // bias index is affine in j (slope -128 floats); base at j=7 keeps offsets >= 0
  const float* bias_base = &rels[((ti_s + 1 - hb) * PP + pi) * PP];

  // ---- S^T = K (Q*scale)^T + bias^T : 8 tiles down the t dimension ----
  v16h bQ = load_b_frag(&Qs[s_me * DD + hb * 16]);  // B: Q columns, hoisted
  v8f Cf[8];
#pragma unroll
  for (int j = 0; j < 8; ++j) {
    // bias[s][t]: pj == r -> 8 contiguous floats, immediate-offset vector load
    v8f c = *(const v8f*)(bias_base + (7 - j) * (2 * PP * PP));
    v16h aK = load_a_frag(&Ks[(16 * j + ln) * DD + hb * 16]);
    Cf[j] = __builtin_amdgcn_wmma_f32_16x16x32_f16(false, aK, false, bQ,
                                                   (short)0, c, false, false);
  }

  // ---- softmax over t (fixed s per lane), exp2 domain, native v_exp_f32 ----
  float m = Cf[0][0];
#pragma unroll
  for (int j = 0; j < 8; ++j)
#pragma unroll
    for (int r = 0; r < 8; ++r) m = fmaxf(m, Cf[j][r]);
  m = fmaxf(m, __shfl_xor(m, 16, 32));

  float ssum = 0.f;
#pragma unroll
  for (int j = 0; j < 8; ++j)
#pragma unroll
    for (int r = 0; r < 8; ++r) {
      float e = __builtin_amdgcn_exp2f(Cf[j][r] - m);  // single v_exp_f32
      Cf[j][r] = e;
      ssum += e;
    }
  ssum += __shfl_xor(ssum, 16, 32);
  const float rinv = __builtin_amdgcn_rcpf(ssum);      // applied in epilogue

  // ---- write unnormalized P row-major: 8 contiguous t per lane -> 1 b128/tile ----
#pragma unroll
  for (int j = 0; j < 8; ++j) {
    v8h ph;
#pragma unroll
    for (int r = 0; r < 8; ++r) ph[r] = (_Float16)Cf[j][r];
    *(v8h*)&Ps[s_me * SS + 16 * j + kb] = ph;
  }

  // redistribute 1/sum to the output-row lane mapping (rows 16w+8hb+r)
  float rv[8];
#pragma unroll
  for (int r = 0; r < 8; ++r) rv[r] = __shfl(rinv, 8 * hb + r, 32);

  // ---- out = (P @ V) * rinv_row : rows [16w,16w+16) (this wave's only) ----
  v16h aP[4];
#pragma unroll
  for (int tc = 0; tc < 4; ++tc)
    aP[tc] = load_a_frag(&Ps[s_me * SS + tc * 32 + hb * 16]);

#pragma unroll
  for (int dt = 0; dt < 2; ++dt) {
    v8f acc = {0.f, 0.f, 0.f, 0.f, 0.f, 0.f, 0.f, 0.f};
#pragma unroll
    for (int tc = 0; tc < 4; ++tc) {
      v16h bV = load_b_frag(&VTs[(dt * 16 + ln) * SS + tc * 32 + hb * 16]);
      acc = __builtin_amdgcn_wmma_f32_16x16x32_f16(false, aP[tc], false, bV,
                                                   (short)0, acc, false, false);
    }
    const int d0 = dt * 16 + ln;
#pragma unroll
    for (int r = 0; r < 8; ++r) {
      int s = 16 * w + hb * 8 + r;
      out[((size_t)b * SS + s) * (HH * DD) + h * DD + d0] = acc[r] * rv[r];
    }
  }
}

extern "C" void kernel_launch(void* const* d_in, const int* in_sizes, int n_in,
                              void* d_out, int out_size, void* d_ws, size_t ws_size,
                              hipStream_t stream) {
  (void)d_ws; (void)ws_size; (void)out_size;
  const float* x = (const float*)d_in[0];
  const float* rel = (const float*)d_in[1];
  float* out = (float*)d_out;
  int B = 1024;
  if (n_in > 0 && in_sizes[0] > 0) B = in_sizes[0] / (SS * CC);
  msa_fused_f16wmma<<<dim3((unsigned)(B * HH)), dim3(256), 0, stream>>>(x, rel, out);
}
